// IN_22351009809136
// MI455X (gfx1250) — compile-verified, matrix operand order checked by reading, not run on hardware
//
#include <hip/hip_runtime.h>
#include <math.h>

typedef __attribute__((ext_vector_type(16))) _Float16 v16h;
typedef __attribute__((ext_vector_type(8)))  _Float16 v8h;
typedef __attribute__((ext_vector_type(4)))  _Float16 v4h;
typedef __attribute__((ext_vector_type(8)))  float    v8f;
typedef __attribute__((ext_vector_type(4)))  float    v4f;

// ---- WMMA fragment helpers (16x16x32 f16, wave32) ----------------------
// 16-bit A layout (ISA 7.12.2): lane L -> m=L&15, hs=L>>4; per lane the 16
// halves are two contiguous runs of 8: K = ko+8*hs+[0..7] and K = ko+16+8*hs+[0..7].
// Each run is 16 bytes and 16B-aligned for our strides -> 2x ds_load_b128.
__device__ __forceinline__ v16h frag_a(const _Float16* buf, int stride,
                                       int mo, int ko, int lane) {
  int m = mo + (lane & 15);
  int off = m * stride + ko + ((lane >> 4) << 3);
  v8h lo = *(const v8h*)(buf + off);
  v8h hi = *(const v8h*)(buf + off + 16);
  return __builtin_shufflevector(lo, hi, 0, 1, 2, 3, 4, 5, 6, 7,
                                         8, 9, 10, 11, 12, 13, 14, 15);
}

// B fragment from an N-major staged buffer BT[n][k] (B[k][n] transposed):
// lane L -> n=L&15, same two contiguous K runs -> wide loads.
__device__ __forceinline__ v16h frag_bt(const _Float16* buf, int stride,
                                        int ko, int no, int lane) {
  int n = no + (lane & 15);
  int off = n * stride + ko + ((lane >> 4) << 3);
  v8h lo = *(const v8h*)(buf + off);
  v8h hi = *(const v8h*)(buf + off + 16);
  return __builtin_shufflevector(lo, hi, 0, 1, 2, 3, 4, 5, 6, 7,
                                         8, 9, 10, 11, 12, 13, 14, 15);
}

#define WMMA_F16(A, B, C) \
  __builtin_amdgcn_wmma_f32_16x16x32_f16(false, (A), false, (B), (short)0, (C), false, false)

// ---- fused interaction-network kernel: one block per batch element -----
__global__ __launch_bounds__(256) void in_net_kernel(
    const float* __restrict__ x,  const float* __restrict__ y,
    const float* __restrict__ W1_pp, const float* __restrict__ b1_pp,
    const float* __restrict__ W2_pp, const float* __restrict__ b2_pp,
    const float* __restrict__ W1_pv, const float* __restrict__ b1_pv,
    const float* __restrict__ W2_pv, const float* __restrict__ b2_pv,
    const float* __restrict__ W1_o,  const float* __restrict__ b1_o,
    const float* __restrict__ W2_o,  const float* __restrict__ b2_o,
    const float* __restrict__ Wc,    const float* __restrict__ bc,
    float* __restrict__ out)
{
  __shared__ __align__(16) unsigned char smem[63360];
  _Float16* XT     = (_Float16*)(smem + 0);      // [64][64]  XT[i][p] = x[b,p,i]
  _Float16* WbufT  = (_Float16*)(smem + 8192);   // [192 n][64 k] B^T of [W1r|W1s|W1pv_x]
  _Float16* Hcat   = (_Float16*)(smem + 8192);   // [64][128] (reuse)
  _Float16* W2catT = (_Float16*)(smem + 24576);  // [32 n][128 k]
  _Float16* W1oT   = (_Float16*)(smem + 8192);   // [64 n][96 k] (reuse)
  _Float16* JL     = (_Float16*)(smem + 20480);  // [64 m][96 k] (reuse)
  _Float16* AppL   = (_Float16*)(smem + 32768);  // [64][64]
  _Float16* BppL   = (_Float16*)(smem + 40960);  // [64][64]
  _Float16* AxL    = (_Float16*)(smem + 49152);  // [64][64]
  _Float16* EbarL  = (_Float16*)(smem + 32768);  // [64][32] (reuse AppL)
  float*    HoL    = (float*)(smem + 36864);     // [64][64] f32
  float*    b1ppL  = (float*)(smem + 57344);     // [64]
  float*    b1pvL  = (float*)(smem + 57600);     // [64]
  float*    b1oL   = (float*)(smem + 57856);     // [64]
  float*    wvecL  = (float*)(smem + 58112);     // [64]  W2_o @ Wc
  float*    ebcL   = (float*)(smem + 58368);     // [32]  59*b2_pp + 14*b2_pv
  float*    colsum = (float*)(smem + 58496);     // [64]
  float*    AyL    = (float*)(smem + 58752);     // [14][64] f32
  float*    psum   = (float*)(smem + 62336);     // [4][64]

  const int t = threadIdx.x;
  const int lane = t & 31;
  const int wid = t >> 5;
  const int b = blockIdx.x;
  const float* xb = x + (size_t)b * 3600;   // [P=60][Np=60]
  const float* yb = y + (size_t)b * 70;     // [S=5][Nv=14]

  // ---- Phase A: zero-init, then load inputs & weights ----
  for (int idx = t; idx < 64 * 64;  idx += 256) XT[idx]    = (_Float16)0.f;
  for (int idx = t; idx < 192 * 64; idx += 256) WbufT[idx] = (_Float16)0.f;
  for (int idx = t; idx < 14 * 64;  idx += 256) AyL[idx]   = 0.f;
  __syncthreads();

  for (int idx = t; idx < 3600; idx += 256) {
    int p = idx / 60, i = idx % 60;
    XT[i * 64 + p] = (_Float16)xb[p * 60 + i];
  }
  for (int idx = t; idx < 7200; idx += 256) {          // W1_pp [120][60]
    int r = idx / 60, h = idx % 60;
    if (r < 60) WbufT[h * 64 + r]              = (_Float16)W1_pp[idx];
    else        WbufT[(64 + h) * 64 + (r - 60)] = (_Float16)W1_pp[idx];
  }
  for (int idx = t; idx < 3600; idx += 256) {          // W1_pv rows 0..59
    int p = idx / 60, h = idx % 60;
    WbufT[(128 + h) * 64 + p] = (_Float16)W1_pv[idx];
  }
  if (t < 64) {
    b1ppL[t] = (t < 60) ? b1_pp[t] : 0.f;
    b1pvL[t] = (t < 60) ? b1_pv[t] : 0.f;
    b1oL[t]  = (t < 60) ? b1_o[t]  : 0.f;
    float wv = 0.f;
    if (t < 60)
      for (int d = 0; d < 24; ++d) wv += W2_o[t * 24 + d] * Wc[d];
    wvecL[t] = wv;
  }
  if (t < 32) ebcL[t] = (t < 20) ? (59.f * b2_pp[t] + 14.f * b2_pv[t]) : 0.f;
  for (int idx = t; idx < 840; idx += 256) {           // Ay[v][h] (tiny)
    int v = idx / 60, h = idx % 60;
    float s = 0.f;
    for (int s5 = 0; s5 < 5; ++s5)
      s += yb[s5 * 14 + v] * W1_pv[(60 + s5) * 60 + h];
    AyL[v * 64 + h] = s;
  }
  __syncthreads();

  // ---- Phase B: GEMM1  [App|Bpp|Ax](64x192) = XT(64x64) @ W(64x192) ----
  {
    int mo = (wid & 3) * 16;
    v16h a0 = frag_a(XT, 64, mo, 0, lane);     // A frags shared by 6 tiles
    v16h a1 = frag_a(XT, 64, mo, 32, lane);
    int ntb = wid >> 2;
    for (int r = 0; r < 6; ++r) {
      int ng = (ntb + 2 * r) * 16;
      v8f acc = {};
      acc = WMMA_F16(a0, frag_bt(WbufT, 64, 0, ng, lane), acc);
      acc = WMMA_F16(a1, frag_bt(WbufT, 64, 32, ng, lane), acc);
      _Float16* dst = (ng < 64) ? AppL : ((ng < 128) ? BppL : AxL);
      int no = ng & 63, n = no + (lane & 15), hs = lane >> 4;
#pragma unroll
      for (int rr = 0; rr < 8; ++rr)
        dst[(mo + rr + 8 * hs) * 64 + n] = (_Float16)acc[rr];
    }
  }
  __syncthreads();

  // ---- Phase C: edge relu-sums -> Hcat ; stage W2catT ----
  for (int idx = t; idx < 32 * 128; idx += 256) W2catT[idx] = (_Float16)0.f;
  __syncthreads();
  for (int idx = t; idx < 1200; idx += 256) {
    int h = idx / 20, d = idx % 20;
    W2catT[d * 128 + h]      = (_Float16)W2_pp[idx];
    W2catT[d * 128 + 64 + h] = (_Float16)W2_pv[idx];
  }
  {
    // register-blocked 4i x 4h micro-kernel, f32 accumulation
    int h0 = (t & 15) * 4;
    int i0 = (t >> 4) * 4;
    float a[4][4], acc[4][4], ax[4][4], acc2[4][4];
#pragma unroll
    for (int ii = 0; ii < 4; ++ii)
#pragma unroll
      for (int hh = 0; hh < 4; ++hh) {
        a[ii][hh]  = (float)AppL[(i0 + ii) * 64 + h0 + hh] + b1ppL[h0 + hh];
        ax[ii][hh] = (float)AxL[(i0 + ii) * 64 + h0 + hh] + b1pvL[h0 + hh];
        acc[ii][hh] = 0.f;
        acc2[ii][hh] = 0.f;
      }
    for (int j = 0; j < 60; ++j) {
      v4h bv = *(const v4h*)(BppL + j * 64 + h0);
      float bf[4];
#pragma unroll
      for (int hh = 0; hh < 4; ++hh) bf[hh] = (float)bv[hh];
#pragma unroll
      for (int ii = 0; ii < 4; ++ii)
#pragma unroll
        for (int hh = 0; hh < 4; ++hh)
          acc[ii][hh] += fmaxf(a[ii][hh] + bf[hh], 0.f);
    }
#pragma unroll
    for (int ii = 0; ii < 4; ++ii) {                    // drop j==i term
      v4h bv = *(const v4h*)(BppL + (i0 + ii) * 64 + h0);
#pragma unroll
      for (int hh = 0; hh < 4; ++hh)
        acc[ii][hh] -= fmaxf(a[ii][hh] + (float)bv[hh], 0.f);
    }
    for (int v = 0; v < 14; ++v) {
      v4f av = *(const v4f*)(AyL + v * 64 + h0);
#pragma unroll
      for (int ii = 0; ii < 4; ++ii)
#pragma unroll
        for (int hh = 0; hh < 4; ++hh)
          acc2[ii][hh] += fmaxf(ax[ii][hh] + av[hh], 0.f);
    }
#pragma unroll
    for (int ii = 0; ii < 4; ++ii) {
      int i = i0 + ii;
      bool ok = (i < 60);
      v4h o1, o2;
#pragma unroll
      for (int hh = 0; hh < 4; ++hh) {
        o1[hh] = ok ? (_Float16)acc[ii][hh]  : (_Float16)0.f;
        o2[hh] = ok ? (_Float16)acc2[ii][hh] : (_Float16)0.f;
      }
      *(v4h*)(Hcat + i * 128 + h0)      = o1;
      *(v4h*)(Hcat + i * 128 + 64 + h0) = o2;
    }
  }
  __syncthreads();

  // ---- Phase D: GEMM2  Ebar(64x32) = Hcat(64x128) @ W2cat(128x32) + ebc ----
  {
    int mo = (wid & 3) * 16, no = (wid >> 2) * 16;   // 8 tiles, 1 per wave
    v8f acc = {};
#pragma unroll
    for (int ks = 0; ks < 128; ks += 32) {
      v16h af = frag_a(Hcat, 128, mo, ks, lane);
      v16h bf = frag_bt(W2catT, 128, ks, no, lane);
      acc = WMMA_F16(af, bf, acc);
    }
    int n = no + (lane & 15), hs = lane >> 4;
#pragma unroll
    for (int rr = 0; rr < 8; ++rr)
      EbarL[(mo + rr + 8 * hs) * 32 + n] = (_Float16)(acc[rr] + ebcL[n]);
  }
  __syncthreads();

  // ---- Phase E: stage W1oT (rows 60..79 -> k=64..83, K padded to 96) and
  //      J = [x_row | Ebar] as an M-major A buffer ----
  for (int idx = t; idx < 64 * 96; idx += 256) W1oT[idx] = (_Float16)0.f;
  for (int idx = t; idx < 64 * 96; idx += 256) JL[idx]   = (_Float16)0.f;
  __syncthreads();
  for (int idx = t; idx < 4800; idx += 256) {
    int r = idx / 60, n = idx % 60;
    int k = (r < 60) ? r : r + 4;
    W1oT[n * 96 + k] = (_Float16)W1_o[idx];
  }
  for (int idx = t; idx < 3600; idx += 256) {          // x part (coalesced)
    int i = idx / 60, k = idx % 60;
    JL[i * 96 + k] = (_Float16)xb[i * 60 + k];
  }
  for (int idx = t; idx < 60 * 20; idx += 256) {       // Ebar part
    int i = idx / 20, d = idx % 20;
    JL[i * 96 + 64 + d] = EbarL[i * 32 + d];
  }
  __syncthreads();

  // ---- Phase F: GEMM3  Ho = relu(J(64x96) @ W1o(96x64) + b1_o) ----
  {
    int mo = (wid & 3) * 16;
    v16h a0 = frag_a(JL, 96, mo, 0, lane);
    v16h a1 = frag_a(JL, 96, mo, 32, lane);
    v16h a2 = frag_a(JL, 96, mo, 64, lane);
    int ntb = wid >> 2;
    for (int r = 0; r < 2; ++r) {
      int no = (ntb + 2 * r) * 16;
      v8f acc = {};
      acc = WMMA_F16(a0, frag_bt(W1oT, 96, 0, no, lane), acc);
      acc = WMMA_F16(a1, frag_bt(W1oT, 96, 32, no, lane), acc);
      acc = WMMA_F16(a2, frag_bt(W1oT, 96, 64, no, lane), acc);
      int n = no + (lane & 15), hs = lane >> 4;
#pragma unroll
      for (int rr = 0; rr < 8; ++rr) {
        int m = mo + rr + 8 * hs;
        float v = fmaxf(acc[rr] + b1oL[n], 0.f);
        HoL[m * 64 + n] = (m < 60) ? v : 0.f;   // mask pad rows (relu(b1)!=0)
      }
    }
  }
  __syncthreads();

  // ---- Phase G: reduce over particles, classifier, sigmoid ----
  {
    int col = t & 63, q = t >> 6;
    float s = 0.f;
    for (int i = q * 16; i < q * 16 + 16; ++i) s += HoL[i * 64 + col];
    psum[q * 64 + col] = s;
  }
  __syncthreads();
  if (t < 64)
    colsum[t] = (psum[t] + psum[64 + t] + psum[128 + t] + psum[192 + t]) * wvecL[t];
  __syncthreads();
  if (t == 0) {
    float tot = 0.f;
    for (int k = 0; k < 64; ++k) tot += colsum[k];
    float c0 = bc[0];
    for (int d = 0; d < 24; ++d) c0 += 60.f * b2_o[d] * Wc[d];
    out[b] = 1.f / (1.f + expf(-(tot + c0)));
  }
}

extern "C" void kernel_launch(void* const* d_in, const int* in_sizes, int n_in,
                              void* d_out, int out_size, void* d_ws, size_t ws_size,
                              hipStream_t stream) {
  (void)n_in; (void)out_size; (void)d_ws; (void)ws_size;
  const float* x     = (const float*)d_in[0];
  const float* y     = (const float*)d_in[1];
  const float* W1_pp = (const float*)d_in[2];
  const float* b1_pp = (const float*)d_in[3];
  const float* W2_pp = (const float*)d_in[4];
  const float* b2_pp = (const float*)d_in[5];
  const float* W1_pv = (const float*)d_in[6];
  const float* b1_pv = (const float*)d_in[7];
  const float* W2_pv = (const float*)d_in[8];
  const float* b2_pv = (const float*)d_in[9];
  const float* W1_o  = (const float*)d_in[10];
  const float* b1_o  = (const float*)d_in[11];
  const float* W2_o  = (const float*)d_in[12];
  const float* b2_o  = (const float*)d_in[13];
  const float* Wc    = (const float*)d_in[14];
  const float* bc    = (const float*)d_in[15];
  float* out = (float*)d_out;

  int B = in_sizes[0] / (60 * 60);   // 1024
  in_net_kernel<<<dim3(B), dim3(256), 0, stream>>>(
      x, y, W1_pp, b1_pp, W2_pp, b2_pp, W1_pv, b1_pv, W2_pv, b2_pv,
      W1_o, b1_o, W2_o, b2_o, Wc, bc, out);
}
